// CapsNet_73761768342110
// MI455X (gfx1250) — matrix operand order repaired and checked by reading, not compiled, not run on hardware
//
#include <hip/hip_runtime.h>

typedef __attribute__((ext_vector_type(16))) _Float16 v16h;
typedef __attribute__((ext_vector_type(8)))  _Float16 v8h;
typedef __attribute__((ext_vector_type(8)))  float    v8f;

// ---------------------------------------------------------------------------
// f32 -> f16 convert (activations)
// ---------------------------------------------------------------------------
__global__ void f32_to_f16_k(const float* __restrict__ s, _Float16* __restrict__ d, int n) {
    int i = blockIdx.x * 256 + threadIdx.x;
    if (i < n) d[i] = (_Float16)s[i];
}

// ---------------------------------------------------------------------------
// Weight pack: f32 [OC][K] -> f16 [OC][Kp], rows zero-padded to Kp (mult of 64).
// Removes all K-guards on the WMMA A side and makes A loads 16B-aligned b128s.
// ---------------------------------------------------------------------------
__global__ void pack_w_k(const float* __restrict__ w, _Float16* __restrict__ wh,
                         int OC, int K, int Kp) {
    int i = blockIdx.x * 256 + threadIdx.x;
    if (i >= OC * Kp) return;
    int oc = i / Kp, kk = i % Kp;
    wh[i] = (kk < K) ? (_Float16)w[(size_t)oc * K + kk] : (_Float16)0.f;
}

// ---------------------------------------------------------------------------
// Implicit-GEMM convolution, WMMA f16 with f32 accumulate.
//   M = OC (A = packed weights, row stride Kp), K = IC*KH*KW, N = B*OH*OW.
// Block = 128 threads (4 waves). Block tile = 64(M) x 32(N).
//   - im2col B tile (32 px x 64 K) gathered cooperatively into LDS once,
//     consumed by all 4 waves as aligned v16h (ds_load_b128) fragments.
//   - Each wave: 16 M-rows, 2 N-tiles -> 2 accumulators; K unrolled x2
//     => 4 independent static v_wmma_f32_16x16x32_f16 per k-iteration.
// OC padded by masking waves with m0 >= OC; Npix is a multiple of 32 at every
// call site; K handled by zero-padded weights + zero-filled B gathers.
// ---------------------------------------------------------------------------
__global__ __launch_bounds__(128) void conv_wmma_k(
    const _Float16* __restrict__ in, const _Float16* __restrict__ wgt,
    const float* __restrict__ bias, float* __restrict__ out,
    int B, int IC, int IH, int IW, int OC, int KH, int KW,
    int OH, int OW, int stride, int pad, int Kp)
{
    const int K    = IC * KH * KW;
    const int khw  = KH * KW;
    const int HWo  = OH * OW;
    const int tid  = threadIdx.x;
    const int wv   = tid >> 5;
    const int lane = tid & 31;
    const int r    = lane & 15;
    const int hi   = lane >> 4;

    const int m0     = blockIdx.x * 64 + wv * 16;
    const bool mvalid = (m0 < OC);
    const int nbase  = blockIdx.y * 32;

    __shared__ __attribute__((aligned(32))) _Float16 s_B[32 * 64];  // [n_local][k_local]
    __shared__ int s_bb[32], s_oy[32], s_ox[32];

    if (tid < 32) {
        int n  = nbase + tid;
        int bb = n / HWo;
        int sp = n % HWo;
        s_bb[tid] = bb;
        s_oy[tid] = sp / OW;
        s_ox[tid] = sp % OW;
    }
    __syncthreads();

    const _Float16* wrow = wgt + (size_t)(m0 + r) * Kp;

    v8f c0 = {0.f,0.f,0.f,0.f,0.f,0.f,0.f,0.f};
    v8f c1 = {0.f,0.f,0.f,0.f,0.f,0.f,0.f,0.f};

    for (int k0 = 0; k0 < Kp; k0 += 64) {
        // ---- cooperative im2col gather: 32 px x 64 K into LDS ----
        #pragma unroll
        for (int t = 0; t < 16; ++t) {
            int idx = tid + t * 128;          // 0..2047
            int nl  = idx >> 6;               // pixel 0..31
            int ko  = idx & 63;               // k offset 0..63
            int kk  = k0 + ko;
            _Float16 v = (_Float16)0.f;
            if (kk < K) {
                int ic  = kk / khw;
                int rem = kk % khw;
                int ky  = rem / KW, kx = rem % KW;
                int iy  = s_oy[nl] * stride + ky - pad;
                int ix  = s_ox[nl] * stride + kx - pad;
                if (iy >= 0 && iy < IH && ix >= 0 && ix < IW)
                    v = in[(((size_t)s_bb[nl] * IC + ic) * IH + iy) * IW + ix];
            }
            s_B[nl * 64 + ko] = v;
        }
        __syncthreads();

        if (mvalid) {
            if (k0 + 64 < Kp) __builtin_prefetch(wrow + k0 + 64, 0, 0);
            #pragma unroll
            for (int sub = 0; sub < 2; ++sub) {
                const int ks = k0 + sub * 32;
                // A fragment: lane r = M-row; a[i] = K ks + hi*16 + i
                v8h alo = *(const v8h*)(wrow + ks + hi * 16);
                v8h ahi = *(const v8h*)(wrow + ks + hi * 16 + 8);
                v16h a;
                #pragma unroll
                for (int i = 0; i < 8; ++i) { a[i] = alo[i]; a[i + 8] = ahi[i]; }
                // B fragments: lane r = N-col; b[i] = K sub*32 + hi*16 + i
                v16h b0 = *(const v16h*)(&s_B[(size_t)r * 64 + sub * 32 + hi * 16]);
                v16h b1 = *(const v16h*)(&s_B[(size_t)(16 + r) * 64 + sub * 32 + hi * 16]);
                c0 = __builtin_amdgcn_wmma_f32_16x16x32_f16(false, a, false, b0, (short)0, c0, false, false);
                c1 = __builtin_amdgcn_wmma_f32_16x16x32_f16(false, a, false, b1, (short)0, c1, false, false);
            }
        }
        __syncthreads();
    }

    if (mvalid) {
        // C/D layout: VGPR v holds M = v + 8*hi, N = lane&15
        #pragma unroll
        for (int v = 0; v < 8; ++v) {
            int oc = m0 + v + 8 * hi;
            float bs = bias[oc];
            int n1 = r, n2 = 16 + r;
            out[(((size_t)s_bb[n1] * OC + oc) * OH + s_oy[n1]) * OW + s_ox[n1]] = c0[v] + bs;
            out[(((size_t)s_bb[n2] * OC + oc) * OH + s_oy[n2]) * OW + s_ox[n2]] = c1[v] + bs;
        }
    }
}

// ---------------------------------------------------------------------------
// BatchNorm batch statistics: one block per channel. x layout [B][C][HW].
// ---------------------------------------------------------------------------
__global__ __launch_bounds__(256) void bn_stats_k(
    const float* __restrict__ x, float* __restrict__ mean, float* __restrict__ var,
    int C, int HW, int Bn)
{
    const int c = blockIdx.x;
    const int N = Bn * HW;
    float s = 0.f, q = 0.f;
    for (int i = threadIdx.x; i < N; i += 256) {
        int b = i / HW, sp = i % HW;
        float v = x[((size_t)b * C + c) * HW + sp];
        s += v; q += v * v;
    }
    __shared__ float rs[256], rq[256];
    rs[threadIdx.x] = s; rq[threadIdx.x] = q;
    __syncthreads();
    for (int st = 128; st > 0; st >>= 1) {
        if (threadIdx.x < st) {
            rs[threadIdx.x] += rs[threadIdx.x + st];
            rq[threadIdx.x] += rq[threadIdx.x + st];
        }
        __syncthreads();
    }
    if (threadIdx.x == 0) {
        float m = rs[0] / (float)N;
        mean[c] = m;
        var[c]  = rq[0] / (float)N - m * m;   // biased variance
    }
}

// mode: 0=none, 1=relu, 2=sigmoid.  Optionally emits f16 copy.
__global__ void bn_apply_k(
    const float* __restrict__ x, const float* __restrict__ mean,
    const float* __restrict__ var, const float* __restrict__ g,
    const float* __restrict__ bt, float* __restrict__ y,
    _Float16* __restrict__ yh, int C, int HW, int total, int mode)
{
    int i = blockIdx.x * 256 + threadIdx.x;
    if (i >= total) return;
    int c = (i / HW) % C;
    float v = g[c] * (x[i] - mean[c]) * rsqrtf(var[c] + 1e-5f) + bt[c];
    if (mode == 1)      v = fmaxf(v, 0.f);
    else if (mode == 2) v = 1.f / (1.f + expf(-v));
    y[i] = v;
    if (yh) yh[i] = (_Float16)v;
}

// ---------------------------------------------------------------------------
// EM routing. One block (256 thr) per spatial position; pose/rr/stats in LDS,
// votes recomputed on the fly: votes[r,o,i*4+m] = sum_j p[r,i*4+j]*W[r%WR,o,j,m]
//
// mode 0 (ConvCaps): pose_src [B][R*16][HW], act_src [B][R][HW]
// mode 1 (ClassCaps): pose_src [B][WR*16][R/WR] with r=(spatial)*WR+k,
//                     act_src flat [B][R]  (reference's (k,h,w) flatten)
// pose_out [B][O*16][HW] (nullable), act_out [B][O][HW]
// ---------------------------------------------------------------------------
template<int R, int O, int WR>
__global__ __launch_bounds__(256) void em_routing_k(
    const float* __restrict__ pose_src, const float* __restrict__ act_src,
    const float* __restrict__ Wm, const float* __restrict__ beta_u,
    const float* __restrict__ beta_a, float* __restrict__ pose_out,
    float* __restrict__ act_out, int HW, int iters, int mode)
{
    constexpr int O16 = O * 16;
    __shared__ float s_p[R * 16];
    __shared__ float s_rr[R * O];
    __shared__ float s_act[R];
    __shared__ float s_mu[O16];
    __shared__ float s_sig[O16];
    __shared__ float s_rps[O];
    __shared__ float s_cost[O];
    __shared__ float s_oact[O];
    __shared__ float s_sc[2];

    const int   pos = blockIdx.x;
    const int   tid = threadIdx.x;
    const int   b   = pos / HW;
    const int   sp  = pos % HW;
    const float eps = 1e-9f;

    if (mode == 0) {
        for (int i = tid; i < R * 16; i += 256)
            s_p[i] = pose_src[((size_t)b * (R * 16) + i) * HW + sp];
        for (int r = tid; r < R; r += 256)
            s_act[r] = act_src[((size_t)b * R + r) * HW + sp];
    } else {
        constexpr int SP = R / WR;
        for (int i = tid; i < R * 16; i += 256) {
            int r = i >> 4, e = i & 15;
            int k = r % WR, so = r / WR;
            s_p[i] = pose_src[((size_t)b * (WR * 16) + k * 16 + e) * SP + so];
        }
        for (int r = tid; r < R; r += 256)
            s_act[r] = act_src[(size_t)b * R + r];
    }
    for (int i = tid; i < R * O; i += 256) s_rr[i] = 1.0f / (float)O;
    __syncthreads();

    const float itmax = fminf((float)iters, 3.0f);
    for (int it = 0; it < iters; ++it) {
        const float temp = 1.0f + (itmax - 1.0f) * (float)it / fmaxf(1.0f, (float)(iters - 1));

        // rr_p = rr * i_act (in place; rr is fully rebuilt by the M-step)
        for (int i = tid; i < R * O; i += 256) s_rr[i] *= s_act[i / O];
        __syncthreads();

        if (tid < O) {
            float s = 0.f;
            #pragma unroll 1
            for (int r = 0; r < R; ++r) s += s_rr[r * O + tid];
            s_rps[tid] = s;
        }
        __syncthreads();

        // o_mean
        for (int idx = tid; idx < O16; idx += 256) {
            int o = idx >> 4, m = idx & 15;
            int i4 = (m >> 2) * 4, mm = m & 3;
            float s = 0.f;
            #pragma unroll 1
            for (int r = 0; r < R; ++r) {
                const float* wp = Wm + ((size_t)(r % WR) * O + o) * 16;
                float v = s_p[r * 16 + i4 + 0] * wp[0 + mm] + s_p[r * 16 + i4 + 1] * wp[4 + mm]
                        + s_p[r * 16 + i4 + 2] * wp[8 + mm] + s_p[r * 16 + i4 + 3] * wp[12 + mm];
                s += s_rr[r * O + o] * v;
            }
            s_mu[idx] = s / (s_rps[o] + eps);
        }
        __syncthreads();

        // o_stdv
        for (int idx = tid; idx < O16; idx += 256) {
            int o = idx >> 4, m = idx & 15;
            int i4 = (m >> 2) * 4, mm = m & 3;
            float mu = s_mu[idx];
            float s = 0.f;
            #pragma unroll 1
            for (int r = 0; r < R; ++r) {
                const float* wp = Wm + ((size_t)(r % WR) * O + o) * 16;
                float v = s_p[r * 16 + i4 + 0] * wp[0 + mm] + s_p[r * 16 + i4 + 1] * wp[4 + mm]
                        + s_p[r * 16 + i4 + 2] * wp[8 + mm] + s_p[r * 16 + i4 + 3] * wp[12 + mm];
                float d = v - mu;
                s += s_rr[r * O + o] * d * d;
            }
            s_sig[idx] = sqrtf(s / (s_rps[o] + eps));
        }
        __syncthreads();

        // o_cost
        if (tid < O) {
            float s = 0.f;
            #pragma unroll
            for (int m = 0; m < 16; ++m) s += beta_u[tid] + logf(s_sig[tid * 16 + m] + eps);
            s_cost[tid] = s * s_rps[tid];
        }
        __syncthreads();
        if (tid == 0) {
            float cm = 0.f;
            for (int o = 0; o < O; ++o) cm += s_cost[o];
            cm /= (float)O;
            float cv = 0.f;
            for (int o = 0; o < O; ++o) { float d = s_cost[o] - cm; cv += d * d; }
            s_sc[0] = cm;
            s_sc[1] = sqrtf(cv / (float)O);
        }
        __syncthreads();
        if (tid < O) {
            float z = temp * (beta_a[tid] + (s_sc[0] - s_cost[tid]) / (s_sc[1] + eps));
            s_oact[tid] = 1.0f / (1.0f + expf(-z));
        }
        __syncthreads();

        // M-step re-weighting
        if (it < iters - 1) {
            for (int idx = tid; idx < R * O; idx += 256) {
                int r = idx / O, o = idx % O;
                const float* wp = Wm + ((size_t)(r % WR) * O + o) * 16;
                float lp = 0.f;
                #pragma unroll
                for (int m = 0; m < 16; ++m) {
                    int i4 = (m >> 2) * 4, mm = m & 3;
                    float v = s_p[r * 16 + i4 + 0] * wp[0 + mm] + s_p[r * 16 + i4 + 1] * wp[4 + mm]
                            + s_p[r * 16 + i4 + 2] * wp[8 + mm] + s_p[r * 16 + i4 + 3] * wp[12 + mm];
                    float mu = s_mu[o * 16 + m], sg = s_sig[o * 16 + m];
                    float d = v - mu;
                    lp += -(d * d) / (2.0f * sg * sg) - logf(sg) - 0.91893853320467274f;
                }
                s_rr[idx] = s_oact[o] * expf(lp);
            }
            __syncthreads();
            for (int r = tid; r < R; r += 256) {
                float s = 0.f;
                for (int o = 0; o < O; ++o) s += s_rr[r * O + o];
                float inv = 1.0f / (s + eps);
                for (int o = 0; o < O; ++o) s_rr[r * O + o] *= inv;
            }
            __syncthreads();
        }
    }

    if (pose_out) {
        for (int idx = tid; idx < O16; idx += 256)
            pose_out[((size_t)b * O16 + idx) * HW + sp] = s_mu[idx];
    }
    if (tid < O)
        act_out[((size_t)b * O + tid) * HW + sp] = s_oact[tid];
}

// ---------------------------------------------------------------------------
// Host launcher
// ---------------------------------------------------------------------------
extern "C" void kernel_launch(void* const* d_in, const int* in_sizes, int n_in,
                              void* d_out, int out_size, void* d_ws, size_t ws_size,
                              hipStream_t stream)
{
    const float* x         = (const float*)d_in[0];
    const float* conv1_w   = (const float*)d_in[1];
    const float* conv1_b   = (const float*)d_in[2];
    const float* bn1_g     = (const float*)d_in[3];
    const float* bn1_b     = (const float*)d_in[4];
    const float* pc_pose_w = (const float*)d_in[5];
    const float* pc_pose_b = (const float*)d_in[6];
    const float* pc_act_w  = (const float*)d_in[7];
    const float* pc_act_b  = (const float*)d_in[8];
    const float* pc_bn_g   = (const float*)d_in[9];
    const float* pc_bn_b   = (const float*)d_in[10];
    const float* cc1_pose_w = (const float*)d_in[11];
    const float* cc1_pose_b = (const float*)d_in[12];
    const float* cc1_act_w  = (const float*)d_in[13];
    const float* cc1_act_b  = (const float*)d_in[14];
    const float* cc1_bn_g   = (const float*)d_in[15];
    const float* cc1_bn_b   = (const float*)d_in[16];
    const float* cc1_W      = (const float*)d_in[17];
    const float* cc1_beta_u = (const float*)d_in[18];
    const float* cc1_beta_a = (const float*)d_in[19];
    const float* cc2_pose_w = (const float*)d_in[20];
    const float* cc2_pose_b = (const float*)d_in[21];
    const float* cc2_act_w  = (const float*)d_in[22];
    const float* cc2_act_b  = (const float*)d_in[23];
    const float* cc2_bn_g   = (const float*)d_in[24];
    const float* cc2_bn_b   = (const float*)d_in[25];
    const float* cc2_W      = (const float*)d_in[26];
    const float* cc2_beta_u = (const float*)d_in[27];
    const float* cc2_beta_a = (const float*)d_in[28];
    const float* cls_W      = (const float*)d_in[29];
    const float* cls_beta_u = (const float*)d_in[30];
    const float* cls_beta_a = (const float*)d_in[31];

    // ---- workspace bump allocator ----
    char*  base = (char*)d_ws;
    size_t off  = 0;
    auto alloc = [&](size_t n, size_t elsz) -> void* {
        void* p = base + off;
        off += ((n * elsz + 255) / 256) * 256;
        return p;
    };

    _Float16* x_h     = (_Float16*)alloc(6272, 2);
    _Float16* wsmall  = (_Float16*)alloc(512 * 64, 2);     // conv1/pc pose/pc act packed (reused)
    _Float16* actw_h  = (_Float16*)alloc(288 * 320, 2);    // cc act weights packed (reused)
    _Float16* bigw_h  = (_Float16*)alloc(4608 * 4608, 2);  // cc pose weights packed (reused)
    _Float16* h_h     = (_Float16*)alloc(50176, 2);
    _Float16* pose1_h = (_Float16*)alloc(802816, 2);
    _Float16* act1_h  = (_Float16*)alloc(50176, 2);
    _Float16* pose2_h = (_Float16*)alloc(147456, 2);
    _Float16* act2_h  = (_Float16*)alloc(9216, 2);

    float* conv1_out = (float*)alloc(50176, 4);          // (8,32,14,14), bn in place -> h
    float* pose1     = (float*)alloc(802816, 4);         // (8,512,14,14)
    float* pcact     = (float*)alloc(50176, 4);          // (8,32,14,14) -> act1
    float* cc1_pose  = (float*)alloc(1327104, 4);        // (8,4608,6,6)
    float* cc1_act   = (float*)alloc(82944, 4);          // (8,288,6,6)
    float* pose2     = (float*)alloc(147456, 4);         // (8,512,6,6)
    float* act2      = (float*)alloc(9216, 4);           // (8,32,6,6)
    float* cc2_pose  = (float*)alloc(589824, 4);         // (8,4608,4,4)
    float* cc2_act   = (float*)alloc(36864, 4);          // (8,288,4,4)
    float* pose3     = (float*)alloc(65536, 4);          // (8,512,4,4)
    float* act3      = (float*)alloc(4096, 4);           // (8,32,4,4)
    float* bmean     = (float*)alloc(512, 4);
    float* bvar      = (float*)alloc(512, 4);

    auto g1 = [](int n) { return dim3((unsigned)((n + 255) / 256)); };
    auto cgrid = [](int OC, int Npix) { return dim3((unsigned)((OC + 63) / 64), (unsigned)(Npix / 32)); };

    // ---------------- Stage 1: conv1 (5x5 s2 p2) + BN + ReLU ----------------
    f32_to_f16_k<<<g1(6272), 256, 0, stream>>>(x, x_h, 6272);
    pack_w_k<<<g1(32 * 64), 256, 0, stream>>>(conv1_w, wsmall, 32, 25, 64);
    conv_wmma_k<<<cgrid(32, 1568), 128, 0, stream>>>(x_h, wsmall, conv1_b, conv1_out,
                                                     8, 1, 28, 28, 32, 5, 5, 14, 14, 2, 2, 64);
    bn_stats_k<<<32, 256, 0, stream>>>(conv1_out, bmean, bvar, 32, 196, 8);
    bn_apply_k<<<g1(50176), 256, 0, stream>>>(conv1_out, bmean, bvar, bn1_g, bn1_b,
                                              conv1_out, h_h, 32, 196, 50176, 1);

    // ---------------- Stage 2: PrimaryCaps ----------------
    pack_w_k<<<g1(512 * 64), 256, 0, stream>>>(pc_pose_w, wsmall, 512, 32, 64);
    conv_wmma_k<<<cgrid(512, 1568), 128, 0, stream>>>(h_h, wsmall, pc_pose_b, pose1,
                                                      8, 32, 14, 14, 512, 1, 1, 14, 14, 1, 0, 64);
    f32_to_f16_k<<<g1(802816), 256, 0, stream>>>(pose1, pose1_h, 802816);
    pack_w_k<<<g1(32 * 64), 256, 0, stream>>>(pc_act_w, wsmall, 32, 32, 64);
    conv_wmma_k<<<cgrid(32, 1568), 128, 0, stream>>>(h_h, wsmall, pc_act_b, pcact,
                                                     8, 32, 14, 14, 32, 1, 1, 14, 14, 1, 0, 64);
    bn_stats_k<<<32, 256, 0, stream>>>(pcact, bmean, bvar, 32, 196, 8);
    bn_apply_k<<<g1(50176), 256, 0, stream>>>(pcact, bmean, bvar, pc_bn_g, pc_bn_b,
                                              pcact, act1_h, 32, 196, 50176, 2);

    // ---------------- Stage 3: ConvCaps1 (3x3 s2) + EM ----------------
    pack_w_k<<<g1(4608 * 4608), 256, 0, stream>>>(cc1_pose_w, bigw_h, 4608, 4608, 4608);
    conv_wmma_k<<<cgrid(4608, 288), 128, 0, stream>>>(pose1_h, bigw_h, cc1_pose_b, cc1_pose,
                                                      8, 512, 14, 14, 4608, 3, 3, 6, 6, 2, 0, 4608);
    pack_w_k<<<g1(288 * 320), 256, 0, stream>>>(cc1_act_w, actw_h, 288, 288, 320);
    conv_wmma_k<<<cgrid(288, 288), 128, 0, stream>>>(act1_h, actw_h, cc1_act_b, cc1_act,
                                                     8, 32, 14, 14, 288, 3, 3, 6, 6, 2, 0, 320);
    bn_stats_k<<<288, 256, 0, stream>>>(cc1_act, bmean, bvar, 288, 36, 8);
    bn_apply_k<<<g1(82944), 256, 0, stream>>>(cc1_act, bmean, bvar, cc1_bn_g, cc1_bn_b,
                                              cc1_act, ((_Float16*)nullptr), 288, 36, 82944, 2);
    em_routing_k<288, 32, 288><<<288, 256, 0, stream>>>(cc1_pose, cc1_act, cc1_W,
                                                        cc1_beta_u, cc1_beta_a,
                                                        pose2, act2, 36, 3, 0);
    f32_to_f16_k<<<g1(147456), 256, 0, stream>>>(pose2, pose2_h, 147456);
    f32_to_f16_k<<<g1(9216), 256, 0, stream>>>(act2, act2_h, 9216);

    // ---------------- Stage 4: ConvCaps2 (3x3 s1) + EM ----------------
    pack_w_k<<<g1(4608 * 4608), 256, 0, stream>>>(cc2_pose_w, bigw_h, 4608, 4608, 4608);
    conv_wmma_k<<<cgrid(4608, 128), 128, 0, stream>>>(pose2_h, bigw_h, cc2_pose_b, cc2_pose,
                                                      8, 512, 6, 6, 4608, 3, 3, 4, 4, 1, 0, 4608);
    pack_w_k<<<g1(288 * 320), 256, 0, stream>>>(cc2_act_w, actw_h, 288, 288, 320);
    conv_wmma_k<<<cgrid(288, 128), 128, 0, stream>>>(act2_h, actw_h, cc2_act_b, cc2_act,
                                                     8, 32, 6, 6, 288, 3, 3, 4, 4, 1, 0, 320);
    bn_stats_k<<<288, 256, 0, stream>>>(cc2_act, bmean, bvar, 288, 16, 8);
    bn_apply_k<<<g1(36864), 256, 0, stream>>>(cc2_act, bmean, bvar, cc2_bn_g, cc2_bn_b,
                                              cc2_act, ((_Float16*)nullptr), 288, 16, 36864, 2);
    em_routing_k<288, 32, 288><<<128, 256, 0, stream>>>(cc2_pose, cc2_act, cc2_W,
                                                        cc2_beta_u, cc2_beta_a,
                                                        pose3, act3, 16, 3, 0);

    // ---------------- Stage 5: ClassCaps EM -> d_out (8,10) ----------------
    em_routing_k<512, 10, 32><<<8, 256, 0, stream>>>(pose3, act3, cls_W,
                                                     cls_beta_u, cls_beta_a,
                                                     (float*)nullptr, (float*)d_out, 1, 3, 1);

    (void)in_sizes; (void)n_in; (void)out_size; (void)ws_size;
}
